// LSTM_3478923510461
// MI455X (gfx1250) — matrix operand order repaired and checked by reading, not compiled, not run on hardware
//
#include <hip/hip_runtime.h>
#include <hip/hip_bf16.h>

// ---------------------------------------------------------------------------
// 2-layer LSTM (B=256,T=256,I=128,H=512) + FC(O=1) for gfx1250 (MI455X).
// Per-step GEMM on v_wmma_f32_16x16x32_bf16. Weights/x pre-packed into WMMA
// fragment-major layouts; per-WG weight fragments staged once into LDS and
// shared by all 8 waves; inner loop software-pipelined (prefetch chunk n+1
// while issuing chunk n's WMMAs).
// ---------------------------------------------------------------------------

typedef __bf16 bf16_t;
typedef __attribute__((ext_vector_type(16))) __bf16 v16bf;
typedef __attribute__((ext_vector_type(8)))  float  v8f;

#define LSTM_B 256
#define LSTM_T 256
#define LSTM_I 128
#define LSTM_H 512
#define LSTM_G 2048  // 4*H

__device__ __forceinline__ float sigmoid_f(float x) {
  return 1.0f / (1.0f + __expf(-x));
}
__device__ __forceinline__ float tanh_f(float x) {
  float ax = fabsf(x);
  float e  = __expf(-2.0f * ax);
  float t  = (1.0f - e) / (1.0f + e);
  return copysignf(t, x);
}

// A-fragment layout (16x32 bf16 per ISA): lane = (m&15) + 16*((kk>>3)&1),
// element j = ((kk>>4)<<3) + (kk&7), where kk = k&31, chunk = k>>5.
// Fragment-major buffer: [mtile][chunk][lane][16] (32B per lane per chunk).
__device__ __forceinline__ size_t afrag_index(int m, int k, int nchunks) {
  const int mt   = m >> 4;
  const int ch   = k >> 5;
  const int kk   = k & 31;
  const int lane = (m & 15) + 16 * ((kk >> 3) & 1);
  const int j    = ((kk >> 4) << 3) + (kk & 7);
  return ((((size_t)mt * nchunks + ch) * 32 + lane) << 4) + j;
}

struct Frag5 { v16bf a, i, f, g, o; };

__device__ __forceinline__ v8f wmma_bf16(v16bf a, v16bf b, v8f c) {
  return __builtin_amdgcn_wmma_f32_16x16x32_bf16(false, a, false, b,
                                                 (short)0, c, false, false);
}

// ---------------------------------------------------------------------------
// One LSTM timestep for one layer. NCHX = input-K chunks (4 or 16).
//   gates[B,4H] = [x_t | h_prev] @ Wcat^T + bias ; pointwise c/h update.
// Grid (B/128, H/16) = (2,32); 256 threads (8 waves). All waves share one
// 16-col n-tile -> weight fragments staged once into LDS per WG. Each wave:
// 16 batch rows x 16 h-cols, all four gates -> pointwise fully in-register.
// ---------------------------------------------------------------------------
template <int NCHX>
__global__ __launch_bounds__(256) void lstm_step_kernel(
    const bf16_t* __restrict__ xfrag,     // [B/16][NCHX][32][16] input frags (step t)
    const bf16_t* __restrict__ hfrag,     // [B/16][16][32][16]   h(t-1) frags
    const bf16_t* __restrict__ Wfrag,     // [4][H/16][NCHX+16][32][16]
    const float*  __restrict__ bias,      // [4H] = b_ih + b_hh
    float*        __restrict__ c_state,   // [B,H] f32, updated in place
    bf16_t*       __restrict__ hfrag_out) // [B/16][16][32][16]   h(t) frags
{
  constexpr int NCH = NCHX + 16;          // total K chunks (20 or 32)
  extern __shared__ __align__(16) bf16_t ldsb[];  // [4][NCH][32][16]

  const int wave  = threadIdx.x >> 5;
  const int lane  = threadIdx.x & 31;
  const int lhalf = lane >> 4;
  const int m0 = blockIdx.x * 128 + wave * 16;          // batch tile base
  const int mt = m0 >> 4;
  const int nt = blockIdx.y;                            // 16-col tile in H
  const int ncol = nt * 16 + (lane & 15);               // output column (lane)

  // ---- stage this n-tile's weight fragments into LDS (shared by 8 waves) ---
  {
    const int tid = threadIdx.x;
#pragma unroll
    for (int q = 0; q < 4; ++q) {
      const uint4* src =
          (const uint4*)(Wfrag + (((size_t)(q * 32 + nt)) * NCH) * 512);
      uint4* dst = (uint4*)(ldsb + (size_t)q * NCH * 512);
      const int n16 = NCH * 64;  // uint4 elements per gate
      for (int i = tid; i < n16; i += 256) dst[i] = src[i];
    }
  }
  __syncthreads();

  // Per-lane base pointers (all subsequent accesses are immediate-offset).
  const bf16_t* ax  = xfrag + (((size_t)mt * NCHX) * 32 + lane) * 16;
  const bf16_t* ah  = hfrag + (((size_t)mt * 16)   * 32 + lane) * 16;
  const bf16_t* lbi = ldsb + (size_t)0 * NCH * 512 + lane * 16;
  const bf16_t* lbf = ldsb + (size_t)1 * NCH * 512 + lane * 16;
  const bf16_t* lbg = ldsb + (size_t)2 * NCH * 512 + lane * 16;
  const bf16_t* lbo = ldsb + (size_t)3 * NCH * 512 + lane * 16;

  // Accumulators initialized with bias (C layout: N = lane&15 for all 8 regs).
  const float vbi = bias[0 * LSTM_H + ncol];
  const float vbf = bias[1 * LSTM_H + ncol];
  const float vbg = bias[2 * LSTM_H + ncol];
  const float vbo = bias[3 * LSTM_H + ncol];
  v8f acc_i = {vbi, vbi, vbi, vbi, vbi, vbi, vbi, vbi};
  v8f acc_f = {vbf, vbf, vbf, vbf, vbf, vbf, vbf, vbf};
  v8f acc_g = {vbg, vbg, vbg, vbg, vbg, vbg, vbg, vbg};
  v8f acc_o = {vbo, vbo, vbo, vbo, vbo, vbo, vbo, vbo};

  auto load_chunk = [&](const bf16_t* abase, int ach, int wch) -> Frag5 {
    Frag5 r;
    r.a = *(const v16bf*)(abase + (size_t)ach * 512);
    r.i = *(const v16bf*)(lbi + (size_t)wch * 512);
    r.f = *(const v16bf*)(lbf + (size_t)wch * 512);
    r.g = *(const v16bf*)(lbg + (size_t)wch * 512);
    r.o = *(const v16bf*)(lbo + (size_t)wch * 512);
    return r;
  };

  // ---- K loop, 2-stage software pipeline (prefetch n+1, issue n) ----------
  Frag5 cur = load_chunk(ax, 0, 0);
#pragma unroll
  for (int ch = 0; ch < NCHX; ++ch) {       // input-projection chunks
    Frag5 nxt;
    if (ch + 1 < NCHX) nxt = load_chunk(ax, ch + 1, ch + 1);
    else               nxt = load_chunk(ah, 0, NCHX);  // seam into recurrent part
    acc_i = wmma_bf16(cur.a, cur.i, acc_i);
    acc_f = wmma_bf16(cur.a, cur.f, acc_f);
    acc_g = wmma_bf16(cur.a, cur.g, acc_g);
    acc_o = wmma_bf16(cur.a, cur.o, acc_o);
    cur = nxt;
  }
#pragma unroll
  for (int ch = 0; ch < 16; ++ch) {         // recurrent chunks
    Frag5 nxt;
    if (ch + 1 < 16) nxt = load_chunk(ah, ch + 1, NCHX + ch + 1);
    acc_i = wmma_bf16(cur.a, cur.i, acc_i);
    acc_f = wmma_bf16(cur.a, cur.f, acc_f);
    acc_g = wmma_bf16(cur.a, cur.g, acc_g);
    acc_o = wmma_bf16(cur.a, cur.o, acc_o);
    if (ch + 1 < 16) cur = nxt;
  }

  // Pointwise LSTM cell update. D layout: reg j, lane L -> M = j + 8*(L>=16),
  // N = L&15. This wave exclusively owns rows m0..m0+15 at column ncol.
  const int kk   = ncol & 31;
  const int chh  = ncol >> 5;
  const int jp   = ((kk >> 4) << 3) + (kk & 7);
  const int lsel = 16 * ((kk >> 3) & 1);
  bf16_t* hob = hfrag_out + ((((size_t)mt * 16 + chh) * 32) << 4) + jp;
#pragma unroll
  for (int j = 0; j < 8; ++j) {
    const int row = m0 + j + lhalf * 8;
    const size_t cidx = (size_t)row * LSTM_H + ncol;
    const float ig = sigmoid_f(acc_i[j]);
    const float fg = sigmoid_f(acc_f[j]);
    const float gg = tanh_f(acc_g[j]);
    const float og = sigmoid_f(acc_o[j]);
    const float cn = fg * c_state[cidx] + ig * gg;
    c_state[cidx] = cn;
    const int lanep = (j + lhalf * 8) + lsel;   // (row&15) + lsel
    hob[(size_t)lanep << 4] = (bf16_t)(og * tanh_f(cn));
  }
}

// ---------------------------------------------------------------------------
// Init kernels (run every launch: harness poisons d_ws).
// ---------------------------------------------------------------------------

// Pack [W_ih | W_hh] (f32) -> bf16 B-fragment-major [4][H/16][NCH][32][16].
// B layout (32x16): lane holds col n = nt*16 + (lane&15), k = ch*32 +
// (lane>=16?8:0) + ((j>>3)<<4) + (j&7)  (matches A-fragment K order).
__global__ void pack_weights_kernel(const float* __restrict__ Wih,
                                    const float* __restrict__ Whh,
                                    bf16_t* __restrict__ Wfrag,
                                    int KI, int KH) {
  const int NCH = (KI + KH) >> 5;
  const size_t total = (size_t)LSTM_G * (KI + KH);
  const size_t idx = (size_t)blockIdx.x * blockDim.x + threadIdx.x;
  if (idx >= total) return;
  const int j    = (int)(idx & 15);
  size_t tmp     = idx >> 4;
  const int lane = (int)(tmp & 31);
  tmp >>= 5;
  const int ch   = (int)(tmp % NCH);
  tmp /= NCH;
  const int nt   = (int)(tmp & 31);
  const int q    = (int)(tmp >> 5);
  const int n    = nt * 16 + (lane & 15);
  const int k    = ch * 32 + ((lane >> 4) << 3) + ((j >> 3) << 4) + (j & 7);
  const int row  = q * LSTM_H + n;
  const float v  = (k < KI) ? Wih[(size_t)row * KI + k]
                            : Whh[(size_t)row * KH + (k - KI)];
  Wfrag[idx] = (bf16_t)v;
}

// Convert x [B,T,I] f32 -> per-step A-fragment bf16 [T][B/16][I/32][32][16].
__global__ void convert_x_kernel(const float* __restrict__ x,
                                 bf16_t* __restrict__ xfrag) {
  const size_t idx = (size_t)blockIdx.x * blockDim.x + threadIdx.x;
  if (idx >= (size_t)LSTM_T * LSTM_B * LSTM_I) return;
  const int j    = (int)(idx & 15);
  const int lane = (int)((idx >> 4) & 31);
  const int ch   = (int)((idx >> 9) & 3);    // I/32 = 4 chunks
  const int mt   = (int)((idx >> 11) & 15);  // B/16 = 16 mtiles
  const int t    = (int)(idx >> 15);
  const int b    = mt * 16 + (lane & 15);
  const int kk   = ((lane >> 4) & 1) * 8 + ((j >> 3) << 4) + (j & 7);
  const int k    = ch * 32 + kk;
  xfrag[idx] = (bf16_t)x[((size_t)b * LSTM_T + t) * LSTM_I + k];
}

__global__ void init_state_kernel(const float* __restrict__ bih,
                                  const float* __restrict__ bhh,
                                  float* __restrict__ bias,
                                  float* __restrict__ c,
                                  bf16_t* __restrict__ h_a,
                                  bf16_t* __restrict__ h_b) {
  const int i = blockIdx.x * blockDim.x + threadIdx.x;
  if (i < LSTM_G) bias[i] = bih[i] + bhh[i];
  if (i < LSTM_B * LSTM_H) {
    c[i]   = 0.0f;
    h_a[i] = (bf16_t)0.0f;
    h_b[i] = (bf16_t)0.0f;
  }
}

// Final FC: out[b] = h2_last[b,:] . fc_w + fc_b  (h2 is in A-fragment layout)
__global__ void fc_kernel(const bf16_t* __restrict__ h2,
                          const float* __restrict__ fcw,
                          const float* __restrict__ fcb,
                          float* __restrict__ out) {
  const int b = blockIdx.x * blockDim.x + threadIdx.x;
  if (b >= LSTM_B) return;
  float s = fcb[0];
  for (int k = 0; k < LSTM_H; ++k)
    s += (float)h2[afrag_index(b, k, 16)] * fcw[k];
  out[b] = s;
}

// ---------------------------------------------------------------------------
extern "C" void kernel_launch(void* const* d_in, const int* in_sizes, int n_in,
                              void* d_out, int out_size, void* d_ws, size_t ws_size,
                              hipStream_t stream) {
  (void)in_sizes; (void)n_in; (void)out_size; (void)ws_size;

  const float* x     = (const float*)d_in[0];
  const float* W_ih0 = (const float*)d_in[1];
  const float* W_hh0 = (const float*)d_in[2];
  const float* b_ih0 = (const float*)d_in[3];
  const float* b_hh0 = (const float*)d_in[4];
  const float* W_ih1 = (const float*)d_in[5];
  const float* W_hh1 = (const float*)d_in[6];
  const float* b_ih1 = (const float*)d_in[7];
  const float* b_hh1 = (const float*)d_in[8];
  const float* fc_w  = (const float*)d_in[9];
  const float* fc_b  = (const float*)d_in[10];
  float* out = (float*)d_out;

  // Workspace carve-up (256B aligned), total ~25 MB (L2-resident).
  char* ws = (char*)d_ws;
  size_t off = 0;
  auto carve = [&](size_t bytes) -> char* {
    char* p = ws + off;
    off = (off + bytes + 255) & ~(size_t)255;
    return p;
  };
  const int K0 = LSTM_I + LSTM_H;   // 640  (20 chunks)
  const int K1 = LSTM_H + LSTM_H;   // 1024 (32 chunks)
  bf16_t* Wfrag0 = (bf16_t*)carve((size_t)LSTM_G * K0 * sizeof(bf16_t));
  bf16_t* Wfrag1 = (bf16_t*)carve((size_t)LSTM_G * K1 * sizeof(bf16_t));
  bf16_t* xfrag  = (bf16_t*)carve((size_t)LSTM_T * LSTM_B * LSTM_I * sizeof(bf16_t));
  float*  bias0  = (float*) carve((size_t)LSTM_G * sizeof(float));
  float*  bias1  = (float*) carve((size_t)LSTM_G * sizeof(float));
  float*  c0     = (float*) carve((size_t)LSTM_B * LSTM_H * sizeof(float));
  float*  c1     = (float*) carve((size_t)LSTM_B * LSTM_H * sizeof(float));
  bf16_t* h0a    = (bf16_t*)carve((size_t)LSTM_B * LSTM_H * sizeof(bf16_t));
  bf16_t* h0b    = (bf16_t*)carve((size_t)LSTM_B * LSTM_H * sizeof(bf16_t));
  bf16_t* h1a    = (bf16_t*)carve((size_t)LSTM_B * LSTM_H * sizeof(bf16_t));
  bf16_t* h1b    = (bf16_t*)carve((size_t)LSTM_B * LSTM_H * sizeof(bf16_t));

  // Allow >64KB dynamic LDS (deterministic host-side attribute, not captured).
  const int lds0 = 4 * ((K0 >> 5)) * 512 * (int)sizeof(bf16_t);  //  80 KB
  const int lds1 = 4 * ((K1 >> 5)) * 512 * (int)sizeof(bf16_t);  // 128 KB
  hipFuncSetAttribute((const void*)&lstm_step_kernel<4>,
                      hipFuncAttributeMaxDynamicSharedMemorySize, lds0);
  hipFuncSetAttribute((const void*)&lstm_step_kernel<16>,
                      hipFuncAttributeMaxDynamicSharedMemorySize, lds1);

  // ---- init (deterministic, every launch) ----
  {
    const size_t tot0 = (size_t)LSTM_G * K0;
    const size_t tot1 = (size_t)LSTM_G * K1;
    const size_t totx = (size_t)LSTM_T * LSTM_B * LSTM_I;
    pack_weights_kernel<<<(unsigned)((tot0 + 255) / 256), 256, 0, stream>>>(
        W_ih0, W_hh0, Wfrag0, LSTM_I, LSTM_H);
    pack_weights_kernel<<<(unsigned)((tot1 + 255) / 256), 256, 0, stream>>>(
        W_ih1, W_hh1, Wfrag1, LSTM_H, LSTM_H);
    convert_x_kernel<<<(unsigned)((totx + 255) / 256), 256, 0, stream>>>(x, xfrag);
    const unsigned sblocks = (LSTM_B * LSTM_H + 255) / 256;
    init_state_kernel<<<sblocks, 256, 0, stream>>>(b_ih0, b_hh0, bias0, c0, h0a, h0b);
    init_state_kernel<<<sblocks, 256, 0, stream>>>(b_ih1, b_hh1, bias1, c1, h1a, h1b);
  }

  // ---- sequential scan: layer0 step t, then layer1 step t ----
  const dim3 grid(LSTM_B / 128, LSTM_H / 16);  // 2 x 32 = 64 WGs
  const size_t xstep = (size_t)LSTM_B * LSTM_I;  // frag elems per timestep
  for (int t = 0; t < LSTM_T; ++t) {
    bf16_t* h0_prev = (t & 1) ? h0b : h0a;
    bf16_t* h0_next = (t & 1) ? h0a : h0b;
    lstm_step_kernel<4><<<grid, 256, lds0, stream>>>(
        xfrag + (size_t)t * xstep, h0_prev, Wfrag0, bias0, c0, h0_next);

    bf16_t* h1_prev = (t & 1) ? h1b : h1a;
    bf16_t* h1_next = (t & 1) ? h1a : h1b;
    lstm_step_kernel<16><<<grid, 256, lds1, stream>>>(
        h0_next, h1_prev, Wfrag1, bias1, c1, h1_next);
  }

  // t = 255 (odd) wrote h1a -> final hidden state of layer 1.
  fc_kernel<<<1, LSTM_B, 0, stream>>>(h1a, fc_w, fc_b, out);
}
